// CascadedSpatialCrossAttention_5093831213721
// MI455X (gfx1250) — matrix operand order repaired and checked by reading, not compiled, not run on hardware
//
#include <hip/hip_runtime.h>

// CascadedSpatialCrossAttention for MI455X (gfx1250, wave32).
// B=8, C=256, G=4 -> c=64 per group, H=W=128, HW=16384.
// Conv3x3 (dominant ~38.7 GFLOP) = implicit GEMM on v_wmma_f32_16x16x32_bf16,
// K reordered as k''=r*64+ic so each K=32 step has a compile-time 3x3 tap;
// input row-tile staged to LDS by the Tensor Data Mover (TENSORcnt) for
// interior rows, A-matrix repacked once per stage to bf16 k''-major.
// Channel GEMM (w1 @ cat) stays fp32 on v_wmma_f32_16x16x4_f32.
// Workspace requirement: ~102 MB of d_ws.

typedef __attribute__((ext_vector_type(16))) __bf16 v16bf;
typedef __attribute__((ext_vector_type(8)))  __bf16 v8bf;
typedef __attribute__((ext_vector_type(8)))  float  v8f;
typedef __attribute__((ext_vector_type(2)))  float  v2f;
typedef __attribute__((ext_vector_type(4)))  unsigned int ui32x4;
typedef __attribute__((ext_vector_type(8)))  int i32x8;
typedef __attribute__((ext_vector_type(4)))  int i32x4;

#ifndef __has_builtin
#define __has_builtin(x) 0
#endif

// Per-pass availability: 1 in the device pass (probe-proven: the round-4
// #error fired ONLY "when compiling for host"), 0 in the host pass where it
// merely prunes the TDM block from host-side parsing of the kernel body.
#if __has_builtin(__builtin_amdgcn_tensor_load_to_lds) && \
    __has_builtin(__builtin_amdgcn_s_wait_tensorcnt)
#define HAS_TDM 1
#else
#define HAS_TDM 0
#endif

// Decisive probe, DEVICE pass only: the TDM builtins must exist for gfx1250.
#if defined(__HIP_DEVICE_COMPILE__) && !HAS_TDM
#error "CDNA5 probe: TDM builtins missing in DEVICE pass"
#endif

#define HW 16384
#define CG 64
#define NB 8
#define EPSV 1e-5f

// LDS tile geometry: [ic][3 rows][128 cols + 4 pad dwords] -> row stride 132,
// channel stride 396 dwords (396 % 64 banks = 12 -> stride-396 gathers spread
// over banks). TDM pad: interval 128 dwords (code 6), amount 4 dwords (code 3).
#define ROWS 132
#define ICS  396

__device__ __forceinline__ float sigm(float x) { return 1.0f / (1.0f + __expf(-x)); }

// ---------------------------------------------------------------------------
// K0: repack conv weights to bf16, K-major order k'' = r*64 + ic
//     (r = kh*3+kw).  w3r[oc][k''] = bf16(w3[oc][ic][kh][kw])
// ---------------------------------------------------------------------------
__global__ __launch_bounds__(256)
void k_repack(const float* __restrict__ w3g, __bf16* __restrict__ w3r)
{
    int oc = blockIdx.x;
    for (int k = threadIdx.x; k < 576; k += 256) {
        int r = k >> 6, ic = k & 63;
        w3r[oc * 576 + k] = (__bf16)w3g[oc * 576 + ic * 9 + r];
    }
}

// ---------------------------------------------------------------------------
// K1: feat = x_group (+ prev stage output), per-(b,c) plane reductions:
//     row means (x_h), col means (x_w), plane mean & rsqrt(var+eps) for _gn(x)
// ---------------------------------------------------------------------------
__global__ __launch_bounds__(256)
void k_stage_in(const float* __restrict__ x, const float* __restrict__ prev_out,
                float* __restrict__ feat, float* __restrict__ rowmean,
                float* __restrict__ colmean, float* __restrict__ gmean,
                float* __restrict__ grstd, int gi)
{
    int plane = blockIdx.x;               // b*64+cc
    int b = plane >> 6, cc = plane & 63;
    const float* src = x + (((size_t)(b * 256 + gi * 64 + cc)) << 14);
    const float* prv = prev_out + (((size_t)(b * 256 + (gi - 1) * 64 + cc)) << 14);
    float* f = feat + (((size_t)plane) << 14);

    __shared__ float rowL[128];
    __shared__ float colL[256];
    __shared__ float red[256];
    __shared__ float red2[256];

    int t = threadIdx.x;
    if (t < 128) rowL[t] = 0.0f;
    __syncthreads();

    int w = t & 127, hb = t >> 7;
    float s = 0.0f, ss = 0.0f, cs = 0.0f;
    for (int j = 0; j < 64; ++j) {
        int h = hb + 2 * j;
        int idx = (h << 7) + w;
        float v = src[idx];
        if (gi > 0) v += prv[idx];
        f[idx] = v;
        s += v; ss += v * v; cs += v;
        atomicAdd(&rowL[h], v);           // ds_add_f32
    }
    colL[t] = cs; red[t] = s; red2[t] = ss;
    __syncthreads();
    for (int st = 128; st > 0; st >>= 1) {
        if (t < st) { red[t] += red[t + st]; red2[t] += red2[t + st]; }
        __syncthreads();
    }
    if (t < 128) {
        rowmean[plane * 128 + t] = rowL[t] * (1.0f / 128.0f);
        colmean[plane * 128 + t] = (colL[t] + colL[t + 128]) * (1.0f / 128.0f);
    }
    if (t == 0) {
        float mu = red[0] * (1.0f / 16384.0f);
        float var = red2[0] * (1.0f / 16384.0f) - mu * mu;
        gmean[plane] = mu;
        grstd[plane] = rsqrtf(var + EPSV);
    }
}

// ---------------------------------------------------------------------------
// K2: hw[b] = w1 (64x64) @ cat[b] (64x256) + b1   via V_WMMA_F32_16X16X4_F32
// ---------------------------------------------------------------------------
__global__ __launch_bounds__(32)
void k_gemm_hw(const float* __restrict__ w1g, const float* __restrict__ b1g,
               const float* __restrict__ rowmean, const float* __restrict__ colmean,
               float* __restrict__ hwbuf)
{
    int bid = blockIdx.x;
    int b  = bid >> 6;
    int mt = (bid >> 4) & 3;
    int nt = bid & 15;
    int lane = threadIdx.x;
    int lm = lane & 15, lh = lane >> 4;

    int m = mt * 16 + lm;
    int l = nt * 16 + lm;
    v8f acc = {};
    for (int k0 = 0; k0 < 64; k0 += 4) {
        int k = k0 + 2 * lh;
        v2f a, bb;
        a[0] = w1g[m * 64 + k];
        a[1] = w1g[m * 64 + k + 1];
        bb[0] = (l < 128) ? rowmean[(b * 64 + k) * 128 + l]
                          : colmean[(b * 64 + k) * 128 + (l - 128)];
        bb[1] = (l < 128) ? rowmean[(b * 64 + k + 1) * 128 + l]
                          : colmean[(b * 64 + k + 1) * 128 + (l - 128)];
        acc = __builtin_amdgcn_wmma_f32_16x16x4_f32(false, a, false, bb,
                                                    (short)0, acc, false, false);
    }
#pragma unroll
    for (int v = 0; v < 8; ++v) {
        int mg = mt * 16 + v + 8 * lh;
        hwbuf[(b * 64 + mg) * 256 + nt * 16 + lm] = acc[v] + b1g[mg];
    }
}

// ---------------------------------------------------------------------------
// K3: plane mean / rstd of gated = feat * sig(xh[h]) * sig(xw[w]); zero x12sum
// ---------------------------------------------------------------------------
__global__ __launch_bounds__(256)
void k_gated_stats(const float* __restrict__ feat, const float* __restrict__ hwbuf,
                   float* __restrict__ g1mean, float* __restrict__ g1rstd,
                   float* __restrict__ x12sum)
{
    int plane = blockIdx.x;
    const float* f = feat + (((size_t)plane) << 14);
    const float* hp = hwbuf + plane * 256;
    __shared__ float red[256];
    __shared__ float red2[256];
    int t = threadIdx.x;
    int w = t & 127, hb = t >> 7;
    float sw = sigm(hp[128 + w]);
    float s = 0.0f, ss = 0.0f;
    for (int j = 0; j < 64; ++j) {
        int h = hb + 2 * j;
        float v = f[(h << 7) + w] * sigm(hp[h]) * sw;
        s += v; ss += v * v;
    }
    red[t] = s; red2[t] = ss;
    __syncthreads();
    for (int st = 128; st > 0; st >>= 1) {
        if (t < st) { red[t] += red[t + st]; red2[t] += red2[t + st]; }
        __syncthreads();
    }
    if (t == 0) {
        float mu = red[0] * (1.0f / 16384.0f);
        float var = red2[0] * (1.0f / 16384.0f) - mu * mu;
        g1mean[plane] = mu;
        g1rstd[plane] = rsqrtf(var + EPSV);
        x12sum[plane] = 0.0f;
    }
}

// ---------------------------------------------------------------------------
// K4: implicit-GEMM 3x3 conv, bf16 WMMA + f32 accumulate, fused x12 epilogue.
// One block per (h, b): M=64 oc x N=128 pixels (full row).  LDS tile
// 64ic x 3rows x 128cols f32 (padded, 101 KB), staged by TDM for interior
// rows, manual staging for h=0/127.  K order k''=r*64+ic (tap constant per
// unrolled step): A = two 16B bf16 global loads, B = stride-396 LDS walk with
// one bounds select per fragment.
// ---------------------------------------------------------------------------
__global__ __launch_bounds__(256)
void k_conv(const float* __restrict__ feat, const __bf16* __restrict__ w3r,
            const float* __restrict__ b3g, const float* __restrict__ gmean,
            const float* __restrict__ grstd, const float* __restrict__ gw,
            const float* __restrict__ gb, float* __restrict__ x12,
            float* __restrict__ x12sum)
{
    const int h = blockIdx.x;
    const int b = blockIdx.y;
    __shared__ float sInF[64 * ICS];      // 101,376 B

    const int t = threadIdx.x;
    const bool interior = (h >= 1) && (h <= 126);

#if HAS_TDM
    if (interior) {
        if (t < 32) {   // wave 0 issues the tensor DMA (EXEC ignored by TDM)
            unsigned long long ga = (unsigned long long)(size_t)
                (feat + (((size_t)(b * 64)) << 14) + ((h - 1) << 7));
            unsigned lds = (unsigned)(size_t)(&sInF[0]);
            ui32x4 g0;
            g0[0] = 1u;                                  // count=1, user D#
            g0[1] = lds;                                 // lds_addr
            g0[2] = (unsigned)(ga & 0xFFFFFFFFull);      // global_addr lo
            g0[3] = (unsigned)(((ga >> 32) & 0x1FFFFFFull) | (2u << 30)); // type=2
            // data_size=4B(2)<<16 | pad_en<<20 | pad_interval(128dw=6)<<22
            // | pad_amount(4dw=3)<<25
            i32x8 g1;
            g1[0] = 0x07920000;
            g1[1] = (int)(128u << 16);   // tensor_dim0 = 128 (X)
            g1[2] = (int)(3u << 16);     // tensor_dim1 = 3   (rows)
            g1[3] = (int)(128u << 16);   // tile_dim0 = 128
            g1[4] = (int)(3u | (64u << 16)); // tile_dim1=3, tile_dim2=64 (ic)
            g1[5] = 128;                 // tensor_dim0_stride = 128 (row)
            g1[6] = (int)(16384u << 16); // tensor_dim1_stride lo16 = 16384 (plane)
            g1[7] = 0;
            i32x4 g2; g2[0] = 64; g2[1] = 0; g2[2] = 0; g2[3] = 0; // tensor_dim2
            i32x4 g3; g3[0] = 0; g3[1] = 0; g3[2] = 0; g3[3] = 0;
#if __clang_major__ >= 23
            i32x8 gz = {};
            __builtin_amdgcn_tensor_load_to_lds(g0, g1, g2, g3, gz, 0);
#else
            __builtin_amdgcn_tensor_load_to_lds(g0, g1, g2, g3, 0);
#endif
            __builtin_amdgcn_s_wait_tensorcnt(0);
        }
    } else
#endif
    {
        // manual staging (border rows h=0 and h=127)
        for (int idx = t; idx < 64 * 384; idx += 256) {
            int ic  = idx / 384;
            int rr  = idx - ic * 384;
            int kh  = rr >> 7;
            int col = rr & 127;
            int hs = h + kh - 1;
            float v = 0.0f;
            if ((unsigned)hs < 128u)
                v = feat[(((size_t)(b * 64 + ic)) << 14) + (hs << 7) + col];
            sInF[ic * ICS + kh * ROWS + col] = v;
        }
    }
    __syncthreads();

    const int wv = t >> 5, lane = t & 31;
    const int lm = lane & 15, lh = lane >> 4;
    const int mt = wv >> 1;               // oc tile 0..3
    const int nb = (wv & 1) * 4;          // first of four pixel tiles
    v8f acc[4] = {};

    const __bf16* arow = w3r + (mt * 16 + lm) * 576;
#pragma unroll
    for (int r = 0; r < 9; ++r) {
        const int kh = r / 3, kw = r % 3;
#pragma unroll
        for (int ih = 0; ih < 2; ++ih) {
            const int kb = (r * 2 + ih) * 32;
            // A fragment: elem j -> k'' = kb + j + 8*(j/8) + 8*lh : two
            // contiguous 8x bf16 chunks -> b128 loads + shuffle.
            const v8bf* pa = (const v8bf*)(arow + kb + 8 * lh);
            v8bf alo = pa[0];
            v8bf ahi = pa[2];
            v16bf af = __builtin_shufflevector(alo, ahi,
                0, 1, 2, 3, 4, 5, 6, 7, 8, 9, 10, 11, 12, 13, 14, 15);
            const int ic0 = ih * 32 + 8 * lh;    // B elem j -> ic = ic0 + j + 8*(j/8)
#pragma unroll
            for (int i = 0; i < 4; ++i) {
                const int n = (nb + i) * 16 + lm;     // pixel column
                const int col = n + kw - 1;
                const bool inb = (unsigned)col < 128u;
                const int base = ic0 * ICS + kh * ROWS + col;
                v16bf bf;
#pragma unroll
                for (int j = 0; j < 16; ++j) {
                    float v = inb ? sInF[base + (j + 8 * (j >> 3)) * ICS] : 0.0f;
                    bf[j] = (__bf16)v;
                }
                acc[i] = __builtin_amdgcn_wmma_f32_16x16x32_bf16(
                    false, af, false, bf, (short)0, acc[i], false, false);
            }
        }
    }

    // epilogue: x12 = conv+b3 + sig(gb)*sig(gn(feat)); plane-sum via shfl+atomic
    float osum[8] = {0.f, 0.f, 0.f, 0.f, 0.f, 0.f, 0.f, 0.f};
#pragma unroll
    for (int i = 0; i < 4; ++i) {
        const int n = (nb + i) * 16 + lm;
#pragma unroll
        for (int v = 0; v < 8; ++v) {
            int oc = mt * 16 + v + 8 * lh;
            float x2 = acc[i][v] + b3g[oc];
            float fv = sInF[oc * ICS + ROWS + n];           // feat[b,oc,h,n]
            float g  = (fv - gmean[b * 64 + oc]) * grstd[b * 64 + oc] * gw[oc] + gb[oc];
            float x12v = x2 + sigm(gb[oc]) * sigm(g);       // g.mean == gb[oc]
            x12[(((size_t)(b * 64 + oc)) << 14) + (h << 7) + n] = x12v;
            osum[v] += x12v;
        }
    }
#pragma unroll
    for (int v = 0; v < 8; ++v) {
        float s = osum[v];
        s += __shfl_xor(s, 1); s += __shfl_xor(s, 2);
        s += __shfl_xor(s, 4); s += __shfl_xor(s, 8);
        if (lm == 0)
            atomicAdd(&x12sum[b * 64 + mt * 16 + v + 8 * lh], s);
    }
}

// ---------------------------------------------------------------------------
// K5: x1 = gn(gated) = (feat*sig(xh)*sig(xw) - g1mean)*g1rstd*gw + gb
// ---------------------------------------------------------------------------
__global__ __launch_bounds__(256)
void k_x1(const float* __restrict__ feat, const float* __restrict__ hwbuf,
          const float* __restrict__ g1mean, const float* __restrict__ g1rstd,
          const float* __restrict__ gw, const float* __restrict__ gb,
          float* __restrict__ x1)
{
    const size_t total = (size_t)NB * CG * HW;
    for (size_t i = (size_t)blockIdx.x * 256 + threadIdx.x; i < total;
         i += (size_t)gridDim.x * 256) {
        int plane = (int)(i >> 14);
        int h = (int)((i >> 7) & 127);
        int w = (int)(i & 127);
        int cc = plane & 63;
        float gated = feat[i] * sigm(hwbuf[plane * 256 + h])
                              * sigm(hwbuf[plane * 256 + 128 + w]);
        x1[i] = (gated - g1mean[plane]) * g1rstd[plane] * gw[cc] + gb[cc];
    }
}

// ---------------------------------------------------------------------------
// K6: x21[b,:] = softmax_c(x12sum/HW); x11 = softmax_c(gb) (batch-invariant)
// ---------------------------------------------------------------------------
__global__ __launch_bounds__(64)
void k_softmax(const float* __restrict__ x12sum, const float* __restrict__ gb,
               float* __restrict__ x11, float* __restrict__ x21)
{
    __shared__ float buf[64];
    int b = blockIdx.x, t = threadIdx.x;

    float v = x12sum[b * 64 + t] * (1.0f / 16384.0f);
    buf[t] = v; __syncthreads();
    for (int s = 32; s > 0; s >>= 1) { if (t < s) buf[t] = fmaxf(buf[t], buf[t + s]); __syncthreads(); }
    float mx = buf[0]; __syncthreads();
    float e = __expf(v - mx);
    buf[t] = e; __syncthreads();
    for (int s = 32; s > 0; s >>= 1) { if (t < s) buf[t] += buf[t + s]; __syncthreads(); }
    x21[b * 64 + t] = e / buf[0];
    __syncthreads();

    if (b == 0) {
        float u = gb[t];
        buf[t] = u; __syncthreads();
        for (int s = 32; s > 0; s >>= 1) { if (t < s) buf[t] = fmaxf(buf[t], buf[t + s]); __syncthreads(); }
        float mx2 = buf[0]; __syncthreads();
        float e2 = __expf(u - mx2);
        buf[t] = e2; __syncthreads();
        for (int s = 32; s > 0; s >>= 1) { if (t < s) buf[t] += buf[t + s]; __syncthreads(); }
        x11[t] = e2 / buf[0];
    }
}

// ---------------------------------------------------------------------------
// K7: weights[b,h,w] = sum_c x11[c]*x12 + x21[b,c]*x1 ; out = feat*sig(weights)
// ---------------------------------------------------------------------------
__global__ __launch_bounds__(128)
void k_out(const float* __restrict__ feat, const float* __restrict__ x1,
           const float* __restrict__ x12, const float* __restrict__ x11,
           const float* __restrict__ x21, float* __restrict__ out, int gi)
{
    int h = blockIdx.x, b = blockIdx.y, w = threadIdx.x;
    size_t pix = ((size_t)h << 7) + w;
    float acc = 0.0f;
    for (int c = 0; c < 64; ++c) {
        size_t idx = (((size_t)(b * 64 + c)) << 14) + pix;
        acc += x11[c] * x12[idx] + x21[b * 64 + c] * x1[idx];
    }
    float sw = sigm(acc);
    for (int c = 0; c < 64; ++c) {
        size_t idx = (((size_t)(b * 64 + c)) << 14) + pix;
        out[(((size_t)(b * 256 + gi * 64 + c)) << 14) + pix] = feat[idx] * sw;
    }
}

// ---------------------------------------------------------------------------
extern "C" void kernel_launch(void* const* d_in, const int* in_sizes, int n_in,
                              void* d_out, int out_size, void* d_ws, size_t ws_size,
                              hipStream_t stream)
{
    (void)in_sizes; (void)n_in; (void)out_size; (void)ws_size;
    const float* x   = (const float*)d_in[0];  // (8,256,128,128)
    const float* w1  = (const float*)d_in[1];  // (4,64,64)
    const float* b1  = (const float*)d_in[2];  // (4,64)
    const float* w3  = (const float*)d_in[3];  // (4,64,64,3,3)
    const float* b3  = (const float*)d_in[4];  // (4,64)
    const float* gnw = (const float*)d_in[5];  // (4,64)
    const float* gnb = (const float*)d_in[6];  // (4,64)
    float* out = (float*)d_out;

    float* ws = (float*)d_ws;
    const size_t P = (size_t)NB * CG * HW;     // 8,388,608 floats per map
    float* feat    = ws;
    float* x1buf   = ws + P;
    float* x12buf  = ws + 2 * P;
    float* rowmean = ws + 3 * P;               // B*64*128
    float* colmean = rowmean + NB * CG * 128;
    float* hwbuf   = colmean + NB * CG * 128;  // B*64*256
    float* gmean   = hwbuf + NB * CG * 256;
    float* grstd   = gmean + NB * CG;
    float* g1mean  = grstd + NB * CG;
    float* g1rstd  = g1mean + NB * CG;
    float* x11v    = g1rstd + NB * CG;         // 64
    float* x21v    = x11v + 64;                // B*64
    float* x12sum  = x21v + NB * CG;           // B*64
    __bf16* w3r    = (__bf16*)(x12sum + NB * CG);  // 64*576 bf16 (16B aligned)

    for (int gi = 0; gi < 4; ++gi) {
        k_repack<<<64, 256, 0, stream>>>(w3 + gi * 64 * 576, w3r);
        k_stage_in<<<512, 256, 0, stream>>>(x, out, feat, rowmean, colmean,
                                            gmean, grstd, gi);
        k_gemm_hw<<<512, 32, 0, stream>>>(w1 + gi * 64 * 64, b1 + gi * 64,
                                          rowmean, colmean, hwbuf);
        k_gated_stats<<<512, 256, 0, stream>>>(feat, hwbuf, g1mean, g1rstd, x12sum);
        k_conv<<<dim3(128, 8), 256, 0, stream>>>(feat, w3r, b3 + gi * 64,
                                                 gmean, grstd, gnw + gi * 64,
                                                 gnb + gi * 64, x12buf, x12sum);
        k_x1<<<8192, 256, 0, stream>>>(feat, hwbuf, g1mean, g1rstd,
                                       gnw + gi * 64, gnb + gi * 64, x1buf);
        k_softmax<<<8, 64, 0, stream>>>(x12sum, gnb + gi * 64, x11v, x21v);
        k_out<<<dim3(128, 8), 128, 0, stream>>>(feat, x1buf, x12buf, x11v, x21v,
                                                out, gi);
    }
}
